// TGLoss_79139067396672
// MI455X (gfx1250) — matrix operand order repaired and checked by reading, not compiled, not run on hardware
//

#include <hip/hip_runtime.h>
#include <hip/hip_bf16.h>
#include <math.h>

// ---------------------------------------------------------------------------
// TGLoss for MI455X (gfx1250, wave32, WMMA).
//   out[0] = loss_gdd  (GDDM scalar)
//   out[1] = disc_loss (mean TSDM discriminability)
//
// Heavy part: Gram matrix X @ X^T (8192x8192x256). Matrix-pipe bound (all
// data L2-resident at 192MB; HBM irrelevant), so we use the bf16x3
// split-precision trick:
//   x = hi + lo (both bf16);  x.y ~= hi.hi + lo.hi + hi.lo   (fp32 accum)
// => 24x V_WMMA_F32_16X16X32_BF16 per 16x16 tile instead of
//    64x V_WMMA_F32_16X16X4_F32, with ~fp32 accuracy.
// B fragments are software-pipelined one 32-K chunk ahead and the next
// column tile is warmed with global_prefetch. Falls back to the f32 WMMA
// path if the workspace can't hold the bf16 copies.
// ---------------------------------------------------------------------------

typedef float  v2f  __attribute__((ext_vector_type(2)));
typedef float  v8f  __attribute__((ext_vector_type(8)));
typedef __bf16 v8bf  __attribute__((ext_vector_type(8)));
typedef __bf16 v16bf __attribute__((ext_vector_type(16)));

#define DFEAT 256      // feature dim (compile-time for full unroll)
#define KSEL  11       // k+1 smallest distances kept per row (K=10)

__device__ __forceinline__ v8f wmma_f32_16x16x4(v2f a, v2f b, v8f c) {
    return __builtin_amdgcn_wmma_f32_16x16x4_f32(false, a, false, b,
                                                 (short)0, c, false, false);
}
__device__ __forceinline__ v8f wmma_bf16(v16bf a, v16bf b, v8f c) {
    return __builtin_amdgcn_wmma_f32_16x16x32_bf16(false, a, false, b,
                                                   (short)0, c, false, false);
}

// A 16x32 bf16 fragment: lanes 0-15 hold K = kb+0..7 & kb+16..23 (row M=ml),
// lanes 16-31 hold K = kb+8..15 & kb+24..31.  base = row*D + 8*half.
__device__ __forceinline__ v16bf load_abf(const __bf16* base, int kb) {
    v8bf p0 = *(const v8bf*)(base + kb);
    v8bf p1 = *(const v8bf*)(base + kb + 16);
    return __builtin_shufflevector(p0, p1, 0, 1, 2, 3, 4, 5, 6, 7,
                                   8, 9, 10, 11, 12, 13, 14, 15);
}
// B 32x16 bf16 fragment: lanes 0-15 hold K = kb+0..7 & kb+16..23 (col N=ml),
// lanes 16-31 hold K = kb+8..15 & kb+24..31.  base = col*D + 8*half.
__device__ __forceinline__ v16bf load_bbf(const __bf16* base, int kb) {
    v8bf p0 = *(const v8bf*)(base + kb);
    v8bf p1 = *(const v8bf*)(base + kb + 16);
    return __builtin_shufflevector(p0, p1, 0, 1, 2, 3, 4, 5, 6, 7,
                                   8, 9, 10, 11, 12, 13, 14, 15);
}

// ---------------------------------------------------------------------------
// Kernel 0: split fp32 -> (hi, lo) bf16 pair.  lo = bf16(x - float(hi)).
// ---------------------------------------------------------------------------
__global__ __launch_bounds__(256) void split_bf16_kernel(const float* __restrict__ x,
                                                         __bf16* __restrict__ hi,
                                                         __bf16* __restrict__ lo) {
    size_t i = (size_t)blockIdx.x * 256 + threadIdx.x;
    float v = x[i];
    __bf16 h = (__bf16)v;
    hi[i] = h;
    lo[i] = (__bf16)(v - (float)h);
}

// ---------------------------------------------------------------------------
// Kernel 1: per-row squared norms of target features. One block per row.
// ---------------------------------------------------------------------------
__global__ __launch_bounds__(256) void sqnorm_kernel(const float* __restrict__ tgt,
                                                     float* __restrict__ sq) {
    __shared__ float red[256];
    const int t = threadIdx.x;
    const float x = tgt[(size_t)blockIdx.x * DFEAT + t];
    red[t] = x * x;
    __syncthreads();
#pragma unroll
    for (int s = 128; s > 0; s >>= 1) {
        if (t < s) red[t] += red[t + s];
        __syncthreads();
    }
    if (t == 0) sq[blockIdx.x] = red[0];
}

// ---------------------------------------------------------------------------
// Kernel 2: GDDM. Single block, 1024 threads.
// out[0] = | (mean(src) - mean(tgt)) . w_gddm + b_gddm |
// ---------------------------------------------------------------------------
__global__ __launch_bounds__(1024) void gddm_kernel(const float* __restrict__ src,
                                                    const float* __restrict__ tgt,
                                                    const float* __restrict__ w_gddm,
                                                    const float* __restrict__ b_gddm,
                                                    float* __restrict__ out, int N) {
    __shared__ float red[1024];
    const int t  = threadIdx.x;
    const int d  = t & (DFEAT - 1);
    const int r0 = t >> 8;              // 0..3
    float ss = 0.f, ts = 0.f;
    for (int n = r0; n < N; n += 4) {
        ss += src[(size_t)n * DFEAT + d];
        ts += tgt[(size_t)n * DFEAT + d];
    }
    red[t] = (ss - ts) * w_gddm[d];
    __syncthreads();
#pragma unroll
    for (int s = 512; s > 0; s >>= 1) {
        if (t < s) red[t] += red[t + s];
        __syncthreads();
    }
    if (t == 0) out[0] = fabsf(red[0] / (float)N + b_gddm[0]);
}

// ---------------------------------------------------------------------------
// Kernel 3: TSDM main kernel (templated on dot-product engine).
// 256 threads = 8 waves; each wave owns a 16-row strip, sweeps N/16 column
// tiles, keeps per-row running sum + top-11 smallest distances.
// ---------------------------------------------------------------------------
template <bool BF16>
__global__ __launch_bounds__(256) void tsdm_kernel(const float* __restrict__ tgt,
                                                   const __bf16* __restrict__ thi,
                                                   const __bf16* __restrict__ tlo,
                                                   const float* __restrict__ sq,
                                                   const float* __restrict__ w_tsdm,
                                                   const float* __restrict__ b_tsdm,
                                                   float* __restrict__ partials,
                                                   int N) {
    __shared__ float ltile[8][16 * 17];   // per-wave 16x16 tile, padded
    __shared__ float ldisc[128];

    const int lane = threadIdx.x & 31;
    const int wave = threadIdx.x >> 5;
    const int half = lane >> 4;
    const int ml   = lane & 15;
    const int row_base = blockIdx.x * 128 + wave * 16;
    float* tp = &ltile[wave][0];

    // Register-resident A fragments for the full D=256.
    v2f   afrag[BF16 ? 1 : 64];
    v16bf ahi[BF16 ? 8 : 1];
    v16bf alo[BF16 ? 8 : 1];
    if constexpr (BF16) {
        const __bf16* ah = thi + (size_t)(row_base + ml) * DFEAT + 8 * half;
        const __bf16* al = tlo + (size_t)(row_base + ml) * DFEAT + 8 * half;
#pragma unroll
        for (int ch = 0; ch < 8; ++ch) {
            ahi[ch] = load_abf(ah, 32 * ch);
            alo[ch] = load_abf(al, 32 * ch);
        }
    } else {
        // f32 A 16x4: lanes 0-15 K=kb+0,1; lanes 16-31 K=kb+2,3.
        const float* arow = tgt + (size_t)(row_base + ml) * DFEAT + 2 * half;
#pragma unroll
        for (int i = 0; i < 64; ++i) afrag[i] = *(const v2f*)(arow + 4 * i);
    }

    float sqm[8];
#pragma unroll
    for (int r = 0; r < 8; ++r) sqm[r] = sq[row_base + r + 8 * half];

    float topk[KSEL];
#pragma unroll
    for (int t = 0; t < KSEL; ++t) topk[t] = 3.0e38f;
    float curmax = 3.0e38f;
    float rowsum = 0.f;

    const int ntiles = N >> 4;
    for (int ct = 0; ct < ntiles; ++ct) {
        const int col_base = ct << 4;
        const float sqn = sq[col_base + ml];

        v8f c0 = {}, c1 = {}, c2 = {};
        if constexpr (BF16) {
            const __bf16* bh = thi + (size_t)(col_base + ml) * DFEAT + 8 * half;
            const __bf16* bl = tlo + (size_t)(col_base + ml) * DFEAT + 8 * half;

            // Warm L0 for the next column tile (global_prefetch_b8).
            if (ct + 1 < ntiles) {
                const __bf16* nh = thi + (size_t)(col_base + 16 + ml) * DFEAT;
                const __bf16* nl = tlo + (size_t)(col_base + 16 + ml) * DFEAT;
                __builtin_prefetch(nh, 0, 3);
                __builtin_prefetch(nh + 128, 0, 3);
                __builtin_prefetch(nl, 0, 3);
                __builtin_prefetch(nl + 128, 0, 3);
            }

            // Software pipeline: B fragments loaded one 32-K chunk ahead.
            v16bf fbh = load_bbf(bh, 0);
            v16bf fbl = load_bbf(bl, 0);
#pragma unroll
            for (int ch = 0; ch < 8; ++ch) {
                v16bf nbh = fbh, nbl = fbl;
                if (ch < 7) {
                    nbh = load_bbf(bh, 32 * (ch + 1));
                    nbl = load_bbf(bl, 32 * (ch + 1));
                }
                c0 = wmma_bf16(ahi[ch], fbh, c0);   // hi.hi
                c1 = wmma_bf16(alo[ch], fbh, c1);   // lo.hi
                c2 = wmma_bf16(ahi[ch], fbl, c2);   // hi.lo
                fbh = nbh;
                fbl = nbl;
            }
        } else {
            const float* bp = tgt + (size_t)(col_base + ml) * DFEAT + 2 * half;
#pragma unroll
            for (int i = 0; i < 64; i += 2) {
                v2f b0 = *(const v2f*)(bp + 4 * i);
                v2f b1 = *(const v2f*)(bp + 4 * i + 4);
                c0 = wmma_f32_16x16x4(afrag[i], b0, c0);
                c1 = wmma_f32_16x16x4(afrag[i + 1], b1, c1);
            }
        }

        // dist(m,n) = sqrt(max(0, |x_m|^2 + |x_n|^2 - 2 x_m.x_n)) -> LDS
#pragma unroll
        for (int r = 0; r < 8; ++r) {
            float dot = c0[r] + c1[r] + c2[r];
            float d2 = sqm[r] + sqn - 2.0f * dot;
            tp[(r + 8 * half) * 17 + ml] = sqrtf(fmaxf(d2, 0.0f));
        }
        asm volatile("s_wait_dscnt 0" ::: "memory");
        __builtin_amdgcn_wave_barrier();

        // Lane maintains row `ml`, columns [half*8, half*8+8).
        const float* rp = &tp[ml * 17 + half * 8];
#pragma unroll
        for (int j = 0; j < 8; ++j) {
            float cand = rp[j];
            rowsum += cand;
            if (cand < curmax) {
                int   mi = 0;
                float mv = topk[0];
#pragma unroll
                for (int t = 1; t < KSEL; ++t)
                    if (topk[t] > mv) { mv = topk[t]; mi = t; }
#pragma unroll
                for (int t = 0; t < KSEL; ++t)
                    if (t == mi) topk[t] = cand;
                mv = topk[0];
#pragma unroll
                for (int t = 1; t < KSEL; ++t) mv = fmaxf(mv, topk[t]);
                curmax = mv;
            }
        }
        asm volatile("s_wait_dscnt 0" ::: "memory");
        __builtin_amdgcn_wave_barrier();
    }

    // Merge the two half-wave partials per row (union of two top-11 sets
    // contains the global top-11).
    if (half) {
#pragma unroll
        for (int t = 0; t < KSEL; ++t) tp[ml * 17 + t] = topk[t];
        tp[ml * 17 + KSEL] = rowsum;
    }
    asm volatile("s_wait_dscnt 0" ::: "memory");
    __builtin_amdgcn_wave_barrier();

    if (!half) {
#pragma unroll
        for (int t2 = 0; t2 < KSEL; ++t2) {
            float cand = tp[ml * 17 + t2];
            if (cand < curmax) {
                int   mi = 0;
                float mv = topk[0];
#pragma unroll
                for (int t = 1; t < KSEL; ++t)
                    if (topk[t] > mv) { mv = topk[t]; mi = t; }
#pragma unroll
                for (int t = 0; t < KSEL; ++t)
                    if (t == mi) topk[t] = cand;
                mv = topk[0];
#pragma unroll
                for (int t = 1; t < KSEL; ++t) mv = fmaxf(mv, topk[t]);
                curmax = mv;
            }
        }
        rowsum += tp[ml * 17 + KSEL];

        float s = 0.f, mn = topk[0];
#pragma unroll
        for (int t = 0; t < KSEL; ++t) { s += topk[t]; mn = fminf(mn, topk[t]); }
        float comp = (s - mn) * (1.0f / (float)(KSEL - 1)); // drop self, /10
        float sep  = rowsum / (float)(N - 1);
        ldisc[wave * 16 + ml] =
            fabsf(w_tsdm[0] * comp + w_tsdm[1] * sep + b_tsdm[0]);
    }
    __syncthreads();
#pragma unroll
    for (int s = 64; s > 0; s >>= 1) {
        if (threadIdx.x < s) ldisc[threadIdx.x] += ldisc[threadIdx.x + s];
        __syncthreads();
    }
    if (threadIdx.x == 0) partials[blockIdx.x] = ldisc[0];
}

// ---------------------------------------------------------------------------
// Kernel 4: deterministic fixed-order reduction of per-block partials.
// ---------------------------------------------------------------------------
__global__ void finalize_kernel(const float* __restrict__ partials, int nparts,
                                float* __restrict__ out, int N) {
    if (threadIdx.x == 0 && blockIdx.x == 0) {
        float s = 0.f;
        for (int i = 0; i < nparts; ++i) s += partials[i];
        out[1] = s / (float)N;
    }
}

extern "C" void kernel_launch(void* const* d_in, const int* in_sizes, int n_in,
                              void* d_out, int out_size, void* d_ws, size_t ws_size,
                              hipStream_t stream) {
    const float* src    = (const float*)d_in[0];
    const float* tgt    = (const float*)d_in[1];
    const float* w_tsdm = (const float*)d_in[2];
    const float* b_tsdm = (const float*)d_in[3];
    const float* w_gddm = (const float*)d_in[4];
    const float* b_gddm = (const float*)d_in[5];

    const int D = in_sizes[4];           // 256
    const int N = in_sizes[1] / D;       // 8192
    float* out      = (float*)d_out;
    float* sq       = (float*)d_ws;      // N floats
    const int nblocks = N / 128;
    float* partials = sq + N;            // nblocks floats

    // bf16 hi/lo copies behind the fp32 scratch (16B aligned)
    uintptr_t p = (uintptr_t)(partials + nblocks);
    p = (p + 15) & ~(uintptr_t)15;
    __bf16* thi = (__bf16*)p;
    __bf16* tlo = thi + (size_t)N * D;
    const size_t need = (size_t)((char*)(tlo + (size_t)N * D) - (char*)d_ws);
    const bool use_bf16 = (ws_size >= need);

    sqnorm_kernel<<<N, 256, 0, stream>>>(tgt, sq);
    gddm_kernel<<<1, 1024, 0, stream>>>(src, tgt, w_gddm, b_gddm, out, N);

    if (use_bf16) {
        split_bf16_kernel<<<N, 256, 0, stream>>>(tgt, thi, tlo);
        tsdm_kernel<true><<<nblocks, 256, 0, stream>>>(tgt, thi, tlo, sq,
                                                       w_tsdm, b_tsdm, partials, N);
    } else {
        tsdm_kernel<false><<<nblocks, 256, 0, stream>>>(tgt, nullptr, nullptr, sq,
                                                        w_tsdm, b_tsdm, partials, N);
    }
    finalize_kernel<<<1, 64, 0, stream>>>(partials, nblocks, out, N);
}
